// PRM_26706106647003
// MI455X (gfx1250) — matrix operand order repaired and checked by reading, not compiled, not run on hardware
//
#include <hip/hip_runtime.h>

typedef _Float16 f16;
typedef __attribute__((ext_vector_type(16))) _Float16 v16h;
typedef __attribute__((ext_vector_type(2)))  _Float16 v2h;
typedef __attribute__((ext_vector_type(8)))  float    v8f;

namespace {

constexpr int SEQ  = 200;   // valid sequence length
constexpr int SP   = 208;   // row-padded (13 * 16)
constexpr int SKP  = 224;   // k-padded for P@V (7 * 32)
constexpr int FDIM = 128;
constexpr int DDIM = 64;
constexpr int DFF  = 128;

constexpr int LDX  = 72;    // stride (f16 elems) for 64-wide row-major (A-side) buffers
constexpr int LDW  = 136;   // stride for 128-wide row-major buffers
constexpr int LDPB = 232;   // P block [16][232]
constexpr int LDSB = 216;   // score block stride (floats)

// B-side buffers use a fragment-native swizzle:
//   element (k, n) at ((k>>4)*NC + n)*16 + (k&15)
// so one lane's 16 fragment halves are 32 contiguous bytes (2x ds_load_b128).

// LDS arena offsets (bytes); lifetimes allow the aliasing noted below.
constexpr int XB_OFF = 0;        // x fp16 [208][72]                      (29,952, persistent)
constexpr int QB_OFF = 29952;    // Q [208][72] | (with KT:) vinb/hb | sb (29,952)
constexpr int KT_OFF = 59904;    // Kt swizzled 4g x 208 x 16             (26,624)
constexpr int VB_OFF = 86528;    // V swizzled 14g x 64 x 16 | posb f16   (28,672)
constexpr int WB_OFF = 115200;   // weight swizzled, max 8g x 64 / 4g x128 (16,384)
constexpr int SB_OFF = 131584;   // score f32 [16][216] | lnbuf | obuf | logits (13,824)
constexpr int PB_OFF = 145408;   // P fp16 [16][232]                      (7,424)
constexpr int BF_OFF = 152832;   // 256 floats: bias/gamma/beta/scratch   (1,024)
constexpr int ARENA  = 153856;

union Frag {
  v16h v;
  f16  h[16];
  unsigned int w[8];
};

// A-matrix 16x32 f16 fragment (row-major source, stride ld).
__device__ inline void load_a_frag(Frag& f, const f16* p, int ld, int m0, int k0, int lane) {
  const int m  = m0 + (lane & 15);
  const int hi = lane >> 4;
  const f16* row = p + m * ld + k0 + hi * 16;
#pragma unroll
  for (int i = 0; i < 4; ++i)
    f.w[i] = *(const unsigned int*)(row + 2 * i);
#pragma unroll
  for (int i = 0; i < 4; ++i)
    f.w[4 + i] = *(const unsigned int*)(row + 8 + 2 * i);
}

// B-matrix 32x16 f16 fragment from swizzled storage: 2x 16B loads.
__device__ inline void load_b_sw(Frag& f, const f16* p, int NC, int k0, int n0, int lane) {
  const int n = n0 + (lane & 15);
  const int g = (k0 >> 4) + (lane >> 4);
  const f16* src = p + (g * NC + n) * 16;
  *(uint4*)&f.w[0] = *(const uint4*)(src);
  *(uint4*)&f.w[4] = *(const uint4*)(src + 8);
}

__device__ inline v8f tile_mm(const f16* A, int lda, int m0,
                              const f16* Bsw, int NC, int n0,
                              int K, int lane) {
  v8f acc = {0.f, 0.f, 0.f, 0.f, 0.f, 0.f, 0.f, 0.f};
  for (int k0 = 0; k0 < K; k0 += 32) {
    Frag fa, fb;
    load_a_frag(fa, A, lda, m0, k0, lane);
    load_b_sw(fb, Bsw, NC, k0, n0, lane);
    acc = __builtin_amdgcn_wmma_f32_16x16x32_f16(false, fa.v, false, fb.v,
                                                 (short)0, acc, false, false);
  }
  return acc;
}

// pack 8 f32 accumulators into 4 dwords of f16 pairs (one 16B store)
__device__ inline uint4 pack8(const v8f& a) {
  uint4 r;
  r.x = __builtin_bit_cast(unsigned int, __builtin_amdgcn_cvt_pkrtz(a[0], a[1]));
  r.y = __builtin_bit_cast(unsigned int, __builtin_amdgcn_cvt_pkrtz(a[2], a[3]));
  r.z = __builtin_bit_cast(unsigned int, __builtin_amdgcn_cvt_pkrtz(a[4], a[5]));
  r.w = __builtin_bit_cast(unsigned int, __builtin_amdgcn_cvt_pkrtz(a[6], a[7]));
  return r;
}

// stage a [rows x cols] f32 weight into swizzled f16 B layout (NC == cols)
__device__ inline void stage_w_sw(f16* dst, const float* src, int rows, int cols, int tid) {
  for (int i = tid; i < rows * cols; i += 256) {
    int r = i / cols, c = i - r * cols;
    dst[((r >> 4) * cols + c) * 16 + (r & 15)] = (f16)src[i];
  }
}
__device__ inline void stage_f(float* dst, const float* src, int n, int tid) {
  for (int i = tid; i < n; i += 256) dst[i] = src[i];
}

__global__ __launch_bounds__(256, 1)
void prm_fused_kernel(const float* __restrict__ vin,
                      const float* __restrict__ W0,  const float* __restrict__ b0,
                      const float* __restrict__ pos_emb,
                      const float* __restrict__ Wq,  const float* __restrict__ Wk,
                      const float* __restrict__ Wv,  const float* __restrict__ Wo,
                      const float* __restrict__ bo,
                      const float* __restrict__ ln1g, const float* __restrict__ ln1b,
                      const float* __restrict__ W1,  const float* __restrict__ b1,
                      const float* __restrict__ W2,  const float* __restrict__ b2,
                      const float* __restrict__ ln2g, const float* __restrict__ ln2b,
                      const float* __restrict__ Ws1, const float* __restrict__ bs1,
                      const float* __restrict__ Ws2, const float* __restrict__ bs2,
                      float* __restrict__ out) {
  __shared__ __align__(16) unsigned char arena[ARENA];
  f16*   xb    = (f16*)(arena + XB_OFF);
  f16*   qb    = (f16*)(arena + QB_OFF);
  f16*   ktb   = (f16*)(arena + KT_OFF);          // swizzled, NC=208, 4 k-groups
  f16*   vb    = (f16*)(arena + VB_OFF);          // swizzled, NC=64, 14 k-groups
  f16*   wb    = (f16*)(arena + WB_OFF);          // swizzled weight
  float* sblk  = (float*)(arena + SB_OFF);
  float* lnbuf = (float*)(arena + SB_OFF);        // aliases sblk (post-softmax)
  f16*   obuf  = (f16*)(arena + SB_OFF + 4096);   // aliases sblk tail (row-major A)
  f16*   pblk  = (f16*)(arena + PB_OFF);          // row-major A [16][232]
  float* biasf = (float*)(arena + BF_OFF);
  f16*   vinb  = (f16*)(arena + QB_OFF);          // [208][136] spans QB+KT (pre-layer)
  f16*   hb    = (f16*)(arena + QB_OFF);          // [208][136] spans QB+KT (FFN)
  f16*   sb    = (f16*)(arena + QB_OFF);          // scoring tanh buffer [208][72]
  f16*   posb  = (f16*)(arena + VB_OFF);          // pos_emb f16 [208][64] (input phase)
  float* lbuf  = (float*)(arena + SB_OFF);        // logits [208]

  const int tid  = threadIdx.x;
  const int lane = tid & 31;
  const int wave = tid >> 5;
  const int hi   = lane >> 4;
  const int nn   = lane & 15;
  const int bidx = blockIdx.x;

  __builtin_prefetch(vin + (size_t)bidx * SEQ * FDIM, 0, 1);

  // ---- stage v_input (fp16), pos_emb (fp16), W0, b0; zero P pad columns ----
  for (int i = tid; i < SP * (FDIM / 4); i += 256) {
    int r = i >> 5, c = (i & 31) * 4;
    float4 v;
    if (r < SEQ) v = *(const float4*)&vin[((size_t)bidx * SEQ + r) * FDIM + c];
    else         v = make_float4(0.f, 0.f, 0.f, 0.f);
    f16* d = vinb + r * LDW + c;
    d[0] = (f16)v.x; d[1] = (f16)v.y; d[2] = (f16)v.z; d[3] = (f16)v.w;
  }
  for (int i = tid; i < SP * (DDIM / 4); i += 256) {
    int r = i >> 4, c = (i & 15) * 4;
    float4 v;
    if (r < SEQ) v = *(const float4*)&pos_emb[r * DDIM + c];
    else         v = make_float4(0.f, 0.f, 0.f, 0.f);
    f16* d = posb + r * DDIM + c;
    d[0] = (f16)v.x; d[1] = (f16)v.y; d[2] = (f16)v.z; d[3] = (f16)v.w;
  }
  for (int i = tid; i < 16 * LDPB; i += 256) pblk[i] = (f16)0.f;  // cols >=208 stay 0
  stage_w_sw(wb, W0, FDIM, DDIM, tid);
  stage_f(biasf, b0, DDIM, tid);
  __syncthreads();

  // ---- input projection: x = vin@W0 + b0 + pos_emb ----
  for (int t = wave; t < 13 * 4; t += 8) {
    int mt = t >> 2, nt = t & 3;
    v8f acc = tile_mm(vinb, LDW, mt * 16, wb, DDIM, nt * 16, FDIM, lane);
    int n = nt * 16 + nn;
#pragma unroll
    for (int r = 0; r < 8; ++r) {
      int m = mt * 16 + r + 8 * hi;
      float val = acc[r] + biasf[n] + (float)posb[m * DDIM + n];
      xb[m * LDX + n] = (m < SEQ) ? (f16)val : (f16)0.f;
    }
  }
  __syncthreads();

  for (int l = 0; l < 2; ++l) {
    const float* wql = Wq + l * DDIM * DDIM;
    const float* wkl = Wk + l * DDIM * DDIM;
    const float* wvl = Wv + l * DDIM * DDIM;
    const float* wol = Wo + l * DDIM * DDIM;

    // zero V pad rows [208,224) (k-group 13); other pads come out zero naturally
    for (int i = tid; i < 16 * DDIM; i += 256) {
      int j = SP + (i >> 6), d = i & 63;
      vb[(13 * DDIM + d) * 16 + (j & 15)] = (f16)0.f;
    }
    __syncthreads();

    // ---- Q = x@Wq ----
    stage_w_sw(wb, wql, DDIM, DDIM, tid);
    __syncthreads();
    for (int t = wave; t < 52; t += 8) {
      int mt = t >> 2, nt = t & 3;
      v8f acc = tile_mm(xb, LDX, mt * 16, wb, DDIM, nt * 16, DDIM, lane);
      int n = nt * 16 + nn;
#pragma unroll
      for (int r = 0; r < 8; ++r) qb[(mt * 16 + r + 8 * hi) * LDX + n] = (f16)acc[r];
    }
    __syncthreads();
    // ---- Kt (swizzled B for scores): element (k=d, n=j) = (x@Wk)[j][d] ----
    stage_w_sw(wb, wkl, DDIM, DDIM, tid);
    __syncthreads();
    for (int t = wave; t < 52; t += 8) {
      int mt = t >> 2, nt = t & 3;
      v8f acc = tile_mm(xb, LDX, mt * 16, wb, DDIM, nt * 16, DDIM, lane);
      int d = nt * 16 + nn;
#pragma unroll
      for (int r = 0; r < 8; ++r) {
        int j = mt * 16 + r + 8 * hi;
        ktb[((d >> 4) * SP + j) * 16 + (d & 15)] = (f16)acc[r];
      }
    }
    __syncthreads();
    // ---- V (swizzled B for P@V): element (k=j, n=d); 8 outputs contiguous -> b128 ----
    stage_w_sw(wb, wvl, DDIM, DDIM, tid);
    __syncthreads();
    for (int t = wave; t < 52; t += 8) {
      int mt = t >> 2, nt = t & 3;
      v8f acc = tile_mm(xb, LDX, mt * 16, wb, DDIM, nt * 16, DDIM, lane);
      int d = nt * 16 + nn;
      *(uint4*)&vb[(mt * DDIM + d) * 16 + 8 * hi] = pack8(acc);
    }
    __syncthreads();

    // ---- attention per 16-row query block (fused out-proj + residual + LN1) ----
    stage_w_sw(wb, wol, DDIM, DDIM, tid);
    stage_f(biasf, bo + l * DDIM, DDIM, tid);
    stage_f(biasf + 128, ln1g + l * DDIM, DDIM, tid);
    stage_f(biasf + 192, ln1b + l * DDIM, DDIM, tid);
    __syncthreads();

    for (int q0 = 0; q0 < SP; q0 += 16) {
      // scores S = Q_blk @ Kt   (13 tiles of 16x16, K=64)
      for (int t = wave; t < 13; t += 8) {
        v8f acc = tile_mm(qb, LDX, q0, ktb, SP, t * 16, DDIM, lane);
        int n = t * 16 + nn;
#pragma unroll
        for (int r = 0; r < 8; ++r) sblk[(r + 8 * hi) * LDSB + n] = acc[r];
      }
      __syncthreads();
      // masked softmax over j<200, scale 1/sqrt(64)=0.125; 16 threads / row
      {
        int row = tid >> 4, sub = tid & 15;
        float mx = -1e30f;
        for (int j = sub; j < SEQ; j += 16) mx = fmaxf(mx, sblk[row * LDSB + j]);
#pragma unroll
        for (int o = 8; o; o >>= 1) mx = fmaxf(mx, __shfl_xor(mx, o));
        float sum = 0.f;
        for (int j = sub; j < SEQ; j += 16) {
          float e = __expf((sblk[row * LDSB + j] - mx) * 0.125f);
          sblk[row * LDSB + j] = e;
          sum += e;
        }
#pragma unroll
        for (int o = 8; o; o >>= 1) sum += __shfl_xor(sum, o);
        float inv = 1.f / sum;
        for (int j = sub; j < SP; j += 16) {
          float e = (j < SEQ) ? sblk[row * LDSB + j] * inv : 0.f;
          pblk[row * LDPB + j] = (f16)e;
        }
      }
      __syncthreads();
      // O_blk = P @ V   (4 tiles, K=224)
      for (int t = wave; t < 4; t += 8) {
        v8f acc = tile_mm(pblk, LDPB, 0, vb, DDIM, t * 16, SKP, lane);
        int n = t * 16 + nn;
#pragma unroll
        for (int r = 0; r < 8; ++r) obuf[(r + 8 * hi) * LDX + n] = (f16)acc[r];
      }
      __syncthreads();
      // out = O@Wo + bo + x  (pre-LN, f32)
      for (int t = wave; t < 4; t += 8) {
        v8f acc = tile_mm(obuf, LDX, 0, wb, DDIM, t * 16, DDIM, lane);
        int n = t * 16 + nn;
#pragma unroll
        for (int r = 0; r < 8; ++r) {
          int ml = r + 8 * hi;
          lnbuf[ml * DDIM + n] = acc[r] + biasf[n] + (float)xb[(q0 + ml) * LDX + n];
        }
      }
      __syncthreads();
      // LN1 (Keras: g*(x-mean)/(std+eps)+b) -> new x rows
      if (tid < 16) {
        int m = q0 + tid;
        float mean = 0.f;
        for (int d = 0; d < DDIM; ++d) mean += lnbuf[tid * DDIM + d];
        mean *= (1.f / DDIM);
        float var = 0.f;
        for (int d = 0; d < DDIM; ++d) {
          float df = lnbuf[tid * DDIM + d] - mean;
          var += df * df;
        }
        float inv = 1.f / (sqrtf(var * (1.f / DDIM)) + 1e-6f);
        for (int d = 0; d < DDIM; ++d) {
          float v = biasf[128 + d] * (lnbuf[tid * DDIM + d] - mean) * inv + biasf[192 + d];
          xb[m * LDX + d] = (m < SEQ) ? (f16)v : (f16)0.f;
        }
      }
      __syncthreads();
    }

    // ---- FFN: h = relu(x@W1 + b1) ----
    stage_w_sw(wb, W1 + l * DDIM * DFF, DDIM, DFF, tid);
    stage_f(biasf, b1 + l * DFF, DFF, tid);
    __syncthreads();
    for (int t = wave; t < 13 * 8; t += 8) {
      int mt = t >> 3, nt = t & 7;
      v8f acc = tile_mm(xb, LDX, mt * 16, wb, DFF, nt * 16, DDIM, lane);
      int n = nt * 16 + nn;
#pragma unroll
      for (int r = 0; r < 8; ++r) {
        int m = mt * 16 + r + 8 * hi;
        hb[m * LDW + n] = (f16)fmaxf(acc[r] + biasf[n], 0.f);
      }
    }
    __syncthreads();
    // ---- x = x + h@W2 + b2  (in-place pre-LN2) ----
    stage_w_sw(wb, W2 + l * DFF * DDIM, DFF, DDIM, tid);
    stage_f(biasf, b2 + l * DDIM, DDIM, tid);
    stage_f(biasf + 128, ln2g + l * DDIM, DDIM, tid);
    stage_f(biasf + 192, ln2b + l * DDIM, DDIM, tid);
    __syncthreads();
    for (int t = wave; t < 52; t += 8) {
      int mt = t >> 2, nt = t & 3;
      v8f acc = tile_mm(hb, LDW, mt * 16, wb, DDIM, nt * 16, DFF, lane);
      int n = nt * 16 + nn;
#pragma unroll
      for (int r = 0; r < 8; ++r) {
        int m = mt * 16 + r + 8 * hi;
        xb[m * LDX + n] = (f16)(acc[r] + biasf[n] + (float)xb[m * LDX + n]);
      }
    }
    __syncthreads();
    // ---- LN2 over all rows, in place ----
    if (tid < SP) {
      int m = tid;
      float mean = 0.f;
      for (int d = 0; d < DDIM; ++d) mean += (float)xb[m * LDX + d];
      mean *= (1.f / DDIM);
      float var = 0.f;
      for (int d = 0; d < DDIM; ++d) {
        float df = (float)xb[m * LDX + d] - mean;
        var += df * df;
      }
      float inv = 1.f / (sqrtf(var * (1.f / DDIM)) + 1e-6f);
      for (int d = 0; d < DDIM; ++d) {
        float v = biasf[128 + d] * ((float)xb[m * LDX + d] - mean) * inv + biasf[192 + d];
        xb[m * LDX + d] = (m < SEQ) ? (f16)v : (f16)0.f;
      }
    }
    __syncthreads();
  }

  // ---- scoring head: s = tanh(x@Ws1 + bs1) ----
  stage_w_sw(wb, Ws1, DDIM, DDIM, tid);
  stage_f(biasf, bs1, DDIM, tid);
  stage_f(biasf + 64, Ws2, DDIM, tid);
  __syncthreads();
  for (int t = wave; t < 52; t += 8) {
    int mt = t >> 2, nt = t & 3;
    v8f acc = tile_mm(xb, LDX, mt * 16, wb, DDIM, nt * 16, DDIM, lane);
    int n = nt * 16 + nn;
#pragma unroll
    for (int r = 0; r < 8; ++r) {
      int m = mt * 16 + r + 8 * hi;
      sb[m * LDX + n] = (f16)tanhf(acc[r] + biasf[n]);
    }
  }
  __syncthreads();
  // logits = s@Ws2 + bs2
  if (tid < SEQ) {
    float s = bs2[0];
    for (int d = 0; d < DDIM; ++d) s += (float)sb[tid * LDX + d] * biasf[64 + d];
    lbuf[tid] = s;
  }
  __syncthreads();
  if (tid == 0) {
    float mx = -1e30f;
    for (int j = 0; j < SEQ; ++j) mx = fmaxf(mx, lbuf[j]);
    float sm = 0.f;
    for (int j = 0; j < SEQ; ++j) sm += __expf(lbuf[j] - mx);
    biasf[248] = mx;
    biasf[249] = 1.f / sm;
  }
  __syncthreads();
  if (tid < SEQ)
    out[(size_t)bidx * SEQ + tid] = __expf(lbuf[tid] - biasf[248]) * biasf[249];
}

} // anonymous namespace

extern "C" void kernel_launch(void* const* d_in, const int* in_sizes, int n_in,
                              void* d_out, int out_size, void* d_ws, size_t ws_size,
                              hipStream_t stream) {
  (void)n_in; (void)d_ws; (void)ws_size; (void)out_size;
  const float* vin     = (const float*)d_in[0];
  // d_in[1] = pos_input (arange broadcast) -> positions are just the row index
  const float* W0      = (const float*)d_in[2];
  const float* b0      = (const float*)d_in[3];
  const float* pos_emb = (const float*)d_in[4];
  const float* Wq      = (const float*)d_in[5];
  const float* Wk      = (const float*)d_in[6];
  const float* Wv      = (const float*)d_in[7];
  const float* Wo      = (const float*)d_in[8];
  const float* bo      = (const float*)d_in[9];
  const float* ln1g    = (const float*)d_in[10];
  const float* ln1b    = (const float*)d_in[11];
  const float* W1      = (const float*)d_in[12];
  const float* b1      = (const float*)d_in[13];
  const float* W2      = (const float*)d_in[14];
  const float* b2      = (const float*)d_in[15];
  const float* ln2g    = (const float*)d_in[16];
  const float* ln2b    = (const float*)d_in[17];
  const float* Ws1     = (const float*)d_in[18];
  const float* bs1     = (const float*)d_in[19];
  const float* Ws2     = (const float*)d_in[20];
  const float* bs2     = (const float*)d_in[21];

  const int Bn = in_sizes[0] / (SEQ * FDIM);  // 2048
  prm_fused_kernel<<<Bn, 256, 0, stream>>>(vin, W0, b0, pos_emb, Wq, Wk, Wv, Wo, bo,
                                           ln1g, ln1b, W1, b1, W2, b2, ln2g, ln2b,
                                           Ws1, bs1, Ws2, bs2, (float*)d_out);
}
